// TensorBase_32409823215674
// MI455X (gfx1250) — matrix operand order repaired and checked by reading, not compiled, not run on hardware
//
#include <hip/hip_runtime.h>
#include <hip/hip_bf16.h>
#include <math.h>

// ---------------- Types for WMMA ----------------
typedef __attribute__((ext_vector_type(16))) _Float16 v16h;
typedef __attribute__((ext_vector_type(8)))  _Float16 v8h;
typedef __attribute__((ext_vector_type(8)))  float    v8f;

// ---------------- Problem constants ----------------
#define NRAYS 2048
#define S     128
#define G     128
#define APPC  27
#define FC    128
#define INK   390            // in_mlpC
#define KC1   13             // ceil(390/32)
#define KC2   4              // 128/32
#define NT    8              // 128/16
#define PADK  424            // feature LDS row stride (halves): 212 dwords -> bank stride 20 (conflict-free b128)
#define HPAD  136            // hidden LDS row stride (halves): 68 dwords -> bank stride 4 (conflict-free b128)
#define NFRAG_W1 (KC1*NT)    // 104
#define NFRAG_W2 (KC2*NT)    // 32
#define NFRAG_W3 (KC2*1)     // 4
#define NFRAG_TOT (NFRAG_W1+NFRAG_W2+NFRAG_W3)  // 140

// LDS union layout (bytes)
#define SM_H1    0
#define SM_H2    8704                  // 32*136*2
#define SM_U     17408                 // union region
#define SM_FEAT  SM_U                  // 32*424*2 = 27136
#define SM_BASE  (SM_U + 27136)        // 32*32*4  = 4096
#define SM_WST   SM_U                  // staged W2|W3 = 36864
#define SM_TOTAL (SM_U + 36864)        // 54272

#define AABB_MIN_F  (-1.5f)
#define AABB_INV    (2.0f/3.0f)
#define NEAR_F      2.0f
#define FAR_F       6.0f
#define DSHIFT      (-10.0f)
#define ALPHA_K     0.78125f           // delta * distance_scale = (4/128)*25

#define WMMA(a,b,c) __builtin_amdgcn_wmma_f32_16x16x32_f16(false,(a),false,(b),(short)0,(c),false,false)

__device__ __forceinline__ v16h make_a(const _Float16* ap)
{
    v8h lo = *(const v8h*)ap;
    v8h hi = *(const v8h*)(ap + 16);
    return __builtin_shufflevector(lo, hi, 0,1,2,3,4,5,6,7,8,9,10,11,12,13,14,15);
}

// =====================================================================
// Pass 0: pre-swizzle W1/W2/W3 (f32 row-major [K][N]) into f16 fragments
// matching the 16-bit B 32x16 WMMA layout:
//   lane l, half j : element W[kc*32 + (l>=16?8:0) + (j<8?j:j+8)][nt*16 + (l&15)]
// Fragment order: W1 frags [kc*8+nt] (0..103), W2 (104..135), W3 (136..139).
// =====================================================================
__global__ __launch_bounds__(256) void swizzle_weights(
    const float* __restrict__ W1, const float* __restrict__ W2,
    const float* __restrict__ W3, _Float16* __restrict__ wf)
{
    int gid = blockIdx.x * blockDim.x + threadIdx.x;
    if (gid >= NFRAG_TOT * 32) return;
    int frag = gid >> 5;
    int lane = gid & 31;
    int kh   = (lane >> 4) * 8;
    int n    = lane & 15;
    _Float16* dst = wf + (size_t)frag * 512 + (size_t)lane * 16;
    #pragma unroll
    for (int j = 0; j < 16; ++j) {
        int kj = (j < 8) ? j : j + 8;
        float v = 0.0f;
        if (frag < NFRAG_W1) {
            int kc = frag >> 3, nt = frag & 7;
            int k = kc * 32 + kh + kj;
            if (k < INK) v = W1[k * FC + nt * 16 + n];
        } else if (frag < NFRAG_W1 + NFRAG_W2) {
            int f = frag - NFRAG_W1;
            int kc = f >> 3, nt = f & 7;
            v = W2[(kc * 32 + kh + kj) * FC + nt * 16 + n];
        } else {
            int kc = frag - (NFRAG_W1 + NFRAG_W2);
            if (n < 3) v = W3[(kc * 32 + kh + kj) * 3 + n];
        }
        dst[j] = (_Float16)v;
    }
}

// -------- trilinear 8-corner gather --------
__device__ __forceinline__ float tri8(const float* __restrict__ v,
                                      const int* o, const float* w)
{
    float s = 0.0f;
    #pragma unroll
    for (int i = 0; i < 8; ++i) s = fmaf(v[o[i]], w[i], s);
    return s;
}

// =====================================================================
// Pass 1: fused sampling + positional encoding + 3-layer MLP (WMMA)
// grid: 2048 rays * 4 sample-blocks; block: 64 threads (2 waves).
// =====================================================================
__global__ __launch_bounds__(64) void fused_render(
    const float* __restrict__ rays_o, const float* __restrict__ rays_d,
    const float* __restrict__ dgrid,  const float* __restrict__ agrid,
    const _Float16* __restrict__ wf,
    const float* __restrict__ b1, const float* __restrict__ b2,
    const float* __restrict__ b3,
    float* __restrict__ ws_alpha, float* __restrict__ ws_rgb)
{
    __shared__ __align__(16) unsigned char smem[SM_TOTAL];
    _Float16* h1p   = (_Float16*)(smem + SM_H1);     // [32][HPAD]
    _Float16* h2p   = (_Float16*)(smem + SM_H2);     // [32][HPAD]
    _Float16* featp = (_Float16*)(smem + SM_FEAT);   // [32][PADK]
    float*    basep = (float*)   (smem + SM_BASE);   // [32][32]
    _Float16* wst   = (_Float16*)(smem + SM_WST);    // staged W2|W3 frags

    const int tid = threadIdx.x;
    const int ray = blockIdx.x >> 2;
    const int s0  = (blockIdx.x & 3) * 32;

    // warm L2/L0 with the first W1 k-chunks while we do the gathers
    {
        const char* p = (const char*)wf + (size_t)tid * 64;
        #pragma unroll
        for (int q = 0; q < 4; ++q)
            __builtin_prefetch(p + (size_t)q * 4096, 0, 1);
    }

    // --- ray direction (uniform per block)
    float dx = rays_d[ray * 3 + 0], dy = rays_d[ray * 3 + 1], dz = rays_d[ray * 3 + 2];
    float rn = rsqrtf(dx * dx + dy * dy + dz * dz);
    dx *= rn; dy *= rn; dz *= rn;
    const float ox = rays_o[ray * 3 + 0], oy = rays_o[ray * 3 + 1], oz = rays_o[ray * 3 + 2];

    // ================= Phase A: grid gathers =================
    {
        const int sl  = tid & 31;
        const int grp = tid >> 5;
        const int s   = s0 + sl;
        const float t = NEAR_F + (FAR_F - NEAR_F) * (float)s / (float)(S - 1);
        const float cx = (ox + dx * t - AABB_MIN_F) * AABB_INV - 1.0f;
        const float cy = (oy + dy * t - AABB_MIN_F) * AABB_INV - 1.0f;
        const float cz = (oz + dz * t - AABB_MIN_F) * AABB_INV - 1.0f;
        const float mask = (cx >= -1.f && cx <= 1.f && cy >= -1.f && cy <= 1.f &&
                            cz >= -1.f && cz <= 1.f) ? 1.0f : 0.0f;

        float xf = fminf(fmaxf((cx + 1.f) * 0.5f * (float)(G - 1), 0.f), (float)(G - 1));
        float yf = fminf(fmaxf((cy + 1.f) * 0.5f * (float)(G - 1), 0.f), (float)(G - 1));
        float zf = fminf(fmaxf((cz + 1.f) * 0.5f * (float)(G - 1), 0.f), (float)(G - 1));
        int x0 = min(max((int)floorf(xf), 0), G - 2);
        int y0 = min(max((int)floorf(yf), 0), G - 2);
        int z0 = min(max((int)floorf(zf), 0), G - 2);
        float fx = xf - (float)x0, fy = yf - (float)y0, fz = zf - (float)z0;

        const int o000 = (z0 * G + y0) * G + x0;
        int offs[8];
        offs[0] = o000;             offs[1] = o000 + 1;
        offs[2] = o000 + G;         offs[3] = o000 + G + 1;
        offs[4] = o000 + G * G;     offs[5] = o000 + G * G + 1;
        offs[6] = o000 + G * G + G; offs[7] = o000 + G * G + G + 1;
        float wgt[8];
        wgt[0] = (1.f - fz) * (1.f - fy) * (1.f - fx);
        wgt[1] = (1.f - fz) * (1.f - fy) * fx;
        wgt[2] = (1.f - fz) * fy * (1.f - fx);
        wgt[3] = (1.f - fz) * fy * fx;
        wgt[4] = fz * (1.f - fy) * (1.f - fx);
        wgt[5] = fz * (1.f - fy) * fx;
        wgt[6] = fz * fy * (1.f - fx);
        wgt[7] = fz * fy * fx;

        if (grp == 0) {
            float sf = tri8(dgrid, offs, wgt) * mask + DSHIFT;
            float sp = (sf > 15.f) ? sf : log1pf(__expf(sf));        // softplus
            ws_alpha[ray * S + s] = 1.0f - __expf(-sp * ALPHA_K);
            basep[sl * 32 + 27] = dx; basep[sl * 32 + 28] = dy; basep[sl * 32 + 29] = dz;
        }
        for (int c = grp; c < APPC; c += 2)
            basep[sl * 32 + c] = tri8(agrid + (size_t)c * G * G * G, offs, wgt) * mask;
    }
    __syncthreads();

    // ================= Phase B: build f16 MLP input tile =================
    // k: [0..26]=app, [27..29]=dir, [30..353]=PE(app) sin|cos,
    //    [354..389]=PE(dir) sin|cos, [390..423]=0 pad
    for (int idx = tid; idx < 32 * PADK; idx += 64) {
        int sl = idx / PADK, k = idx - sl * PADK;
        float v = 0.0f;
        if (k < 30) {
            v = basep[sl * 32 + k];
        } else if (k < 354) {
            int j = k - 30; int hb = (j >= 162); int q = hb ? j - 162 : j;
            float a = basep[sl * 32 + q / 6] * (float)(1 << (q % 6));
            v = hb ? __cosf(a) : __sinf(a);
        } else if (k < 390) {
            int j = k - 354; int hb = (j >= 18); int q = hb ? j - 18 : j;
            float a = basep[sl * 32 + 27 + q / 6] * (float)(1 << (q % 6));
            v = hb ? __cosf(a) : __sinf(a);
        }
        featp[sl * PADK + k] = (_Float16)v;
    }
    __syncthreads();

    // ================= WMMA MLP =================
    const int lane = tid & 31;
    const int wv   = tid >> 5;                // wave id -> 16-row tile
    const int mrow = wv * 16 + (lane & 15);   // A row this lane supplies
    const int aoff = (lane >> 4) * 8;         // K sub-offset per A layout
    const int mb   = (lane >> 4) * 8;         // C/D row base per lane half
    const int nb   = lane & 15;               // C/D column within n-tile

    // ---- Layer 1: [16x416] x [416x128], B double-buffered from global ----
    // Two passes of 4 n-tiles each to bound register pressure; the kc loop
    // is fully unrolled so the ping-pong buffers resolve via register
    // renaming (no v_mov chains), while loads stay one chunk ahead.
    #pragma unroll
    for (int half = 0; half < 2; ++half) {
        const _Float16* wbase = wf + (size_t)lane * 16 + (size_t)(half * 4) * 512;
        v8f acc[4] = {};
        v16h bcur[4];
        #pragma unroll
        for (int q = 0; q < 4; ++q)
            bcur[q] = *(const v16h*)(wbase + (size_t)q * 512);
        #pragma unroll
        for (int kc = 0; kc < KC1 - 1; ++kc) {
            v16h a = make_a(featp + mrow * PADK + kc * 32 + aoff);
            v16h bnext[4];
            #pragma unroll
            for (int q = 0; q < 4; ++q)   // prefetch next k-chunk's fragments
                bnext[q] = *(const v16h*)(wbase + ((size_t)(kc + 1) * NT + q) * 512);
            #pragma unroll
            for (int q = 0; q < 4; ++q)
                acc[q] = WMMA(a, bcur[q], acc[q]);
            #pragma unroll
            for (int q = 0; q < 4; ++q) bcur[q] = bnext[q];
        }
        {   // last k-chunk
            v16h a = make_a(featp + mrow * PADK + (KC1 - 1) * 32 + aoff);
            #pragma unroll
            for (int q = 0; q < 4; ++q)
                acc[q] = WMMA(a, bcur[q], acc[q]);
        }
        #pragma unroll
        for (int q = 0; q < 4; ++q) {
            int nt = half * 4 + q;
            float bias = b1[nt * 16 + nb];
            #pragma unroll
            for (int j = 0; j < 8; ++j)
                h1p[(wv * 16 + mb + j) * HPAD + nt * 16 + nb] =
                    (_Float16)fmaxf(acc[q][j] + bias, 0.0f);
        }
    }
    __syncthreads();   // all feat reads done; h1 complete

    // ---- Stage W2|W3 fragments into LDS (overwrites feat region) ----
    // 36864 B = 2304 x 16 B, async DMA to LDS, tracked by ASYNCcnt.
    {
        const unsigned long long src =
            (unsigned long long)(const char*)(wf + (size_t)NFRAG_W1 * 512);
        const unsigned int dst0 = (unsigned int)(size_t)(void*)wst;
        for (int i = tid; i < 2304; i += 64) {
            unsigned int dst = dst0 + (unsigned int)i * 16u;
            unsigned long long ga = src + (unsigned long long)i * 16ull;
            asm volatile("global_load_async_to_lds_b128 %0, %1, off"
                         :: "v"(dst), "v"(ga) : "memory");
        }
        asm volatile("s_wait_asynccnt 0x0" ::: "memory");
    }
    __syncthreads();   // staged weights visible to both waves

    // ---- Layer 2: [16x128] x [128x128], B from LDS ----
    v8f acc2[NT];
    #pragma unroll
    for (int nt = 0; nt < NT; ++nt) acc2[nt] = (v8f){};
    #pragma unroll
    for (int kc = 0; kc < KC2; ++kc) {
        v16h a = make_a(h1p + mrow * HPAD + kc * 32 + aoff);
        #pragma unroll
        for (int nt = 0; nt < NT; ++nt) {
            v16h b = *(const v16h*)(wst + (size_t)(kc * NT + nt) * 512 + (size_t)lane * 16);
            acc2[nt] = WMMA(a, b, acc2[nt]);
        }
    }
    #pragma unroll
    for (int nt = 0; nt < NT; ++nt) {
        float bias = b2[nt * 16 + nb];
        #pragma unroll
        for (int j = 0; j < 8; ++j)
            h2p[(wv * 16 + mb + j) * HPAD + nt * 16 + nb] =
                (_Float16)fmaxf(acc2[nt][j] + bias, 0.0f);
    }
    __syncthreads();

    // ---- Layer 3: [16x128] x [128x16] (cols 0..2 valid), B from LDS ----
    const _Float16* wst3 = wst + (size_t)NFRAG_W2 * 512;
    v8f acc3 = {};
    #pragma unroll
    for (int kc = 0; kc < KC2; ++kc) {
        v16h a = make_a(h2p + mrow * HPAD + kc * 32 + aoff);
        v16h b = *(const v16h*)(wst3 + (size_t)kc * 512 + (size_t)lane * 16);
        acc3 = WMMA(a, b, acc3);
    }
    if (nb < 3) {
        float bias = b3[nb];
        #pragma unroll
        for (int j = 0; j < 8; ++j) {
            float v = acc3[j] + bias;
            float r = 1.0f / (1.0f + __expf(-v));        // sigmoid
            int gs = ray * S + s0 + wv * 16 + mb + j;
            ws_rgb[(size_t)gs * 3 + nb] = r;
        }
    }
}

// =====================================================================
// Pass 2: per-ray transmittance scan + white-background composite
// =====================================================================
__global__ __launch_bounds__(256) void composite(
    const float* __restrict__ ws_alpha, const float* __restrict__ ws_rgb,
    float* __restrict__ out)
{
    int r = blockIdx.x * blockDim.x + threadIdx.x;
    if (r >= NRAYS) return;
    float T = 1.0f, acc = 0.0f, cr = 0.0f, cg = 0.0f, cb = 0.0f;
    for (int s = 0; s < S; ++s) {
        float a = ws_alpha[r * S + s];
        float w = a * T;
        const float* p = ws_rgb + (size_t)(r * S + s) * 3;
        cr = fmaf(w, p[0], cr);
        cg = fmaf(w, p[1], cg);
        cb = fmaf(w, p[2], cb);
        acc += w;
        T *= (1.0f - a + 1e-10f);
    }
    float bg = 1.0f - acc;
    out[r * 3 + 0] = fminf(fmaxf(cr + bg, 0.0f), 1.0f);
    out[r * 3 + 1] = fminf(fmaxf(cg + bg, 0.0f), 1.0f);
    out[r * 3 + 2] = fminf(fmaxf(cb + bg, 0.0f), 1.0f);
}

// =====================================================================
extern "C" void kernel_launch(void* const* d_in, const int* in_sizes, int n_in,
                              void* d_out, int out_size, void* d_ws, size_t ws_size,
                              hipStream_t stream)
{
    const float* rays_o = (const float*)d_in[0];
    const float* rays_d = (const float*)d_in[1];
    const float* dgrid  = (const float*)d_in[2];
    const float* agrid  = (const float*)d_in[3];
    const float* W1     = (const float*)d_in[4];
    const float* b1     = (const float*)d_in[5];
    const float* W2     = (const float*)d_in[6];
    const float* b2     = (const float*)d_in[7];
    const float* W3     = (const float*)d_in[8];
    const float* b3     = (const float*)d_in[9];
    // n_samples (d_in[10]) fixed at 128 per setup_inputs

    _Float16* wf       = (_Float16*)d_ws;                              // 140 KiB swizzled weights
    float*    ws_alpha = (float*)((char*)d_ws + (size_t)NFRAG_TOT * 1024);
    float*    ws_rgb   = ws_alpha + (size_t)NRAYS * S;

    swizzle_weights<<<(NFRAG_TOT * 32 + 255) / 256, 256, 0, stream>>>(W1, W2, W3, wf);
    fused_render<<<NRAYS * (S / 32), 64, 0, stream>>>(rays_o, rays_d, dgrid, agrid,
                                                      wf, b1, b2, b3, ws_alpha, ws_rgb);
    composite<<<(NRAYS + 255) / 256, 256, 0, stream>>>(ws_alpha, ws_rgb, (float*)d_out);
}
